// GNN3_37426345017685
// MI455X (gfx1250) — matrix operand order repaired and checked by reading, not compiled, not run on hardware
//
#include <hip/hip_runtime.h>
#include <hip/hip_fp16.h>

#define NN 100000
#define EE 1600000
#define BBG 512
#define HH 128
#define EPSV 1e-5f

typedef __attribute__((ext_vector_type(16))) _Float16 v16h;
typedef __attribute__((ext_vector_type(8)))  _Float16 v8h;
typedef __attribute__((ext_vector_type(8)))  float    v8f;

// ---------- small utility kernels ----------
__global__ void k_fill(float* p, float v, int n) {
    int t = blockIdx.x * blockDim.x + threadIdx.x;
    if (t < n) p[t] = v;
}

__global__ void k_f32h(const float* __restrict__ in, _Float16* __restrict__ out, int n) {
    int t = blockIdx.x * blockDim.x + threadIdx.x;
    if (t < n) out[t] = (_Float16)in[t];
}

// transpose+convert 128x128 fp32 weight (K x N) -> f16 (N x K)
__global__ void k_wtrans(const float* __restrict__ W, _Float16* __restrict__ Wt) {
    int t = blockIdx.x * blockDim.x + threadIdx.x;
    if (t < HH * HH) {
        int n = t >> 7, k = t & 127;
        Wt[(n << 7) + k] = (_Float16)W[(k << 7) + n];
    }
}

__global__ void k_count(const int* __restrict__ batch, float* __restrict__ cnt, int n) {
    int t = blockIdx.x * blockDim.x + threadIdx.x;
    if (t < n) atomicAdd(&cnt[batch[t]], 1.0f);
}

__global__ void k_deg(const int* __restrict__ dst, float* __restrict__ deg, int e) {
    int t = blockIdx.x * blockDim.x + threadIdx.x;
    if (t < e) atomicAdd(&deg[dst[t]], 1.0f);
}

__global__ void k_rsqrt_ip(float* p, int n) {
    int t = blockIdx.x * blockDim.x + threadIdx.x;
    if (t < n) p[t] = rsqrtf(p[t]);
}

// ---------- GraphNorm ----------
__global__ void k_gn_sum(const float* __restrict__ x, const int* __restrict__ batch,
                         float* __restrict__ sums, int n, int C, int relu) {
    int t = blockIdx.x * blockDim.x + threadIdx.x;
    if (t >= n * C) return;
    int node = t / C, c = t - node * C;
    float v = x[t];
    if (relu) v = fmaxf(v, 0.0f);
    atomicAdd(&sums[batch[node] * C + c], v);
}

__global__ void k_gn_div(float* __restrict__ s, const float* __restrict__ cnt, int total, int C) {
    int t = blockIdx.x * blockDim.x + threadIdx.x;
    if (t < total) s[t] /= fmaxf(cnt[t / C], 1.0f);
}

__global__ void k_gn_var(const float* __restrict__ x, const int* __restrict__ batch,
                         const float* __restrict__ mean, const float* __restrict__ a,
                         float* __restrict__ vs, int n, int C, int relu) {
    int t = blockIdx.x * blockDim.x + threadIdx.x;
    if (t >= n * C) return;
    int node = t / C, c = t - node * C;
    int b = batch[node];
    float v = x[t];
    if (relu) v = fmaxf(v, 0.0f);
    float sub = v - a[c] * mean[b * C + c];
    atomicAdd(&vs[b * C + c], sub * sub);
}

__global__ void k_gn_norm(const float* __restrict__ x, const int* __restrict__ batch,
                          const float* __restrict__ mean, const float* __restrict__ var,
                          const float* __restrict__ a, const float* __restrict__ g,
                          const float* __restrict__ bb, float* __restrict__ outF,
                          _Float16* __restrict__ outH, int n, int C, int relu) {
    int t = blockIdx.x * blockDim.x + threadIdx.x;
    if (t >= n * C) return;
    int node = t / C, c = t - node * C;
    int b = batch[node];
    float v = x[t];
    if (relu) v = fmaxf(v, 0.0f);
    float sub = v - a[c] * mean[b * C + c];
    float y = g[c] * sub * rsqrtf(var[b * C + c] + EPSV) + bb[c];
    if (outF) outF[t] = y;
    if (outH) outH[t] = (_Float16)y;
}

// ---------- conv1 (F_IN=4 -> 128) ----------
__global__ void k_conv1(const float* __restrict__ x4, const float* __restrict__ W,
                        float* __restrict__ out, int n) {
    __shared__ float xs[4];
    int node = blockIdx.x;
    if (node >= n) return;
    if (threadIdx.x < 4) xs[threadIdx.x] = x4[node * 4 + threadIdx.x];
    __syncthreads();
    int c = threadIdx.x;
    float acc = xs[0] * W[c] + xs[1] * W[128 + c] + xs[2] * W[256 + c] + xs[3] * W[384 + c];
    out[(size_t)node * HH + c] = acc;
}

// ---------- GCN aggregation ----------
__global__ void k_conv_init(const float* __restrict__ xw, const float* __restrict__ dinv,
                            const float* __restrict__ bias, float* __restrict__ acc, int n) {
    int t = blockIdx.x * blockDim.x + threadIdx.x;
    if (t >= n * HH) return;
    int node = t >> 7, c = t & 127;
    float di = dinv[node];
    acc[t] = bias[c] + di * di * xw[t];   // self-loop term + bias
}

__global__ void k_edge_scatter(const float* __restrict__ xw, const int* __restrict__ src,
                               const int* __restrict__ dst, const float* __restrict__ dinv,
                               float* __restrict__ acc, int e) {
    int tid = blockIdx.x * blockDim.x + threadIdx.x;
    int eid = tid >> 5, lane = tid & 31;
    if (eid >= e) return;
    int s = src[eid], d = dst[eid];
    float nrm = dinv[s] * dinv[d];
    const float4 v = ((const float4*)(xw + (size_t)s * HH))[lane];
    float* ap = acc + (size_t)d * HH + lane * 4;
    atomicAdd(ap + 0, nrm * v.x);
    atomicAdd(ap + 1, nrm * v.y);
    atomicAdd(ap + 2, nrm * v.z);
    atomicAdd(ap + 3, nrm * v.w);
}

// ---------- WMMA GEMM: out[M,128] = act(A_f16[M,128] @ W + bias) ----------
// Wt is N-major f16 (Wt[n*128+k] = W[k,n]). Weights staged in LDS (32 KB).
// Each wave owns one 16-row tile, 8 col-tiles x 4 K-chunks = 32 wmma ops.
// All 8 B-fragments of a K-chunk are preloaded into distinct registers so the
// 16 ds_load_b128 issue as a clause and the 8 WMMAs pipeline behind them
// instead of hitting s_wait_dscnt 0 per-op.
__global__ __launch_bounds__(256) void k_gemm_wmma(
    const _Float16* __restrict__ A, const _Float16* __restrict__ Wt,
    const float* __restrict__ bias, float* __restrict__ outF,
    _Float16* __restrict__ outH, int M, int relu)
{
    __shared__ alignas(32) _Float16 wlds[HH * HH];
    {
        const uint4* sp = (const uint4*)Wt;
        uint4* dp = (uint4*)wlds;
#pragma unroll
        for (int i = 0; i < 8; ++i)
            dp[threadIdx.x + 256 * i] = sp[threadIdx.x + 256 * i];
    }
    __syncthreads();

    const int wave = threadIdx.x >> 5;
    const int lane = threadIdx.x & 31;
    const int tile = blockIdx.x * 8 + wave;
    if (tile * 16 >= M) return;        // wave-uniform: EXEC stays all-1s in live waves
    const int hw = lane >> 4;          // half-wave select
    const int mr = lane & 15;          // row-in-tile (A) and col-in-tile (B/C)
    const int row0 = tile * 16;

    v8f c[8];
#pragma unroll
    for (int t = 0; t < 8; ++t) {
#pragma unroll
        for (int r = 0; r < 8; ++r) c[t][r] = 0.0f;
    }

    const _Float16* arow = A + (size_t)(row0 + mr) * HH;
    const _Float16* bbase = &wlds[mr * HH + hw * 16];
#pragma unroll
    for (int kc = 0; kc < 4; ++kc) {
        // A frag (16x32 f16): lane holds K = kc*32 + hw*16 + [0..15]
        const _Float16* ap = arow + kc * 32 + hw * 16;
        v16h a = *(const v16h*)(ap);

        // B frags (32x16 each): lane n = nt*16+mr, K = kc*32 + hw*16 + [0..15]
        v16h bf[8];
#pragma unroll
        for (int nt = 0; nt < 8; ++nt)
            bf[nt] = *(const v16h*)(bbase + nt * 16 * HH + kc * 32);
#pragma unroll
        for (int nt = 0; nt < 8; ++nt)
            c[nt] = __builtin_amdgcn_wmma_f32_16x16x32_f16(
                false, a, false, bf[nt], (short)0, c[nt], false, false);
    }

    // C layout: VGPR r -> row row0 + r + 8*hw, col nt*16 + mr
#pragma unroll
    for (int nt = 0; nt < 8; ++nt) {
        const int coln = nt * 16 + mr;
        const float bv = bias ? bias[coln] : 0.0f;
#pragma unroll
        for (int r = 0; r < 8; ++r) {
            const int rown = row0 + r + hw * 8;
            float v = c[nt][r] + bv;
            if (relu) v = fmaxf(v, 0.0f);
            if (outF) outF[(size_t)rown * HH + coln] = v;
            if (outH) outH[(size_t)rown * HH + coln] = (_Float16)v;
        }
    }
}

// ---------- vector dot (gate head / final head) ----------
__global__ void k_dot_h(const _Float16* __restrict__ X, const float* __restrict__ w,
                        const float* __restrict__ b, float* __restrict__ out, int n) {
    int t = blockIdx.x * blockDim.x + threadIdx.x;
    if (t >= n) return;
    const _Float16* xr = X + (size_t)t * HH;
    float acc = 0.0f;
#pragma unroll 4
    for (int k = 0; k < HH; ++k) acc += (float)xr[k] * w[k];
    out[t] = acc + b[0];
}

// ---------- attention softmax-pool ----------
__device__ __forceinline__ unsigned enc_f(float f) {
    unsigned u = __float_as_uint(f);
    return u ^ ((u >> 31) ? 0xFFFFFFFFu : 0x80000000u);
}
__device__ __forceinline__ float dec_f(unsigned u) {
    u ^= ((u >> 31) ? 0x80000000u : 0xFFFFFFFFu);
    return __uint_as_float(u);
}

__global__ void k_segmax(const float* __restrict__ g, const int* __restrict__ batch,
                         unsigned* __restrict__ menc, int n) {
    int t = blockIdx.x * blockDim.x + threadIdx.x;
    if (t < n) atomicMax(&menc[batch[t]], enc_f(g[t]));
}

__global__ void k_expsum(float* __restrict__ g, const int* __restrict__ batch,
                         const unsigned* __restrict__ menc, float* __restrict__ z, int n) {
    int t = blockIdx.x * blockDim.x + threadIdx.x;
    if (t >= n) return;
    int b = batch[t];
    float e = __expf(g[t] - dec_f(menc[b]));
    g[t] = e;
    atomicAdd(&z[b], e);
}

__global__ void k_pool(const float* __restrict__ e, const float* __restrict__ z,
                       const int* __restrict__ batch, const float* __restrict__ x,
                       float* __restrict__ gp, int n) {
    int tid = blockIdx.x * blockDim.x + threadIdx.x;
    int node = tid >> 5, lane = tid & 31;
    if (node >= n) return;
    int b = batch[node];
    float alpha = e[node] / z[b];
    const float4 v = ((const float4*)(x + (size_t)node * HH))[lane];
    float* gpp = gp + (size_t)b * HH + lane * 4;
    atomicAdd(gpp + 0, alpha * v.x);
    atomicAdd(gpp + 1, alpha * v.y);
    atomicAdd(gpp + 2, alpha * v.z);
    atomicAdd(gpp + 3, alpha * v.w);
}

// ---------- host ----------
extern "C" void kernel_launch(void* const* d_in, const int* in_sizes, int n_in,
                              void* d_out, int out_size, void* d_ws, size_t ws_size,
                              hipStream_t stream) {
    (void)in_sizes; (void)n_in; (void)out_size; (void)ws_size;
    const float* x_in    = (const float*)d_in[0];
    const int*   src     = (const int*)d_in[1];
    const int*   dst     = (const int*)d_in[2];
    const int*   batch   = (const int*)d_in[3];
    const float* conv1_w = (const float*)d_in[4];
    const float* conv1_b = (const float*)d_in[5];
    const float* conv2_w = (const float*)d_in[6];
    const float* conv2_b = (const float*)d_in[7];
    const float* conv3_w = (const float*)d_in[8];
    const float* conv3_b = (const float*)d_in[9];
    const float* gate_w1 = (const float*)d_in[10];
    const float* gate_b1 = (const float*)d_in[11];
    const float* gate_w2 = (const float*)d_in[12];
    const float* gate_b2 = (const float*)d_in[13];
    const float* gate_w3 = (const float*)d_in[14];
    const float* gate_b3 = (const float*)d_in[15];
    const float* lin1_w  = (const float*)d_in[16];
    const float* lin1_b  = (const float*)d_in[17];
    const float* lin2_w  = (const float*)d_in[18];
    const float* lin2_b  = (const float*)d_in[19];
    const float* last_w  = (const float*)d_in[20];
    const float* last_b  = (const float*)d_in[21];
    const float* gn_g[4] = {(const float*)d_in[22], (const float*)d_in[25],
                            (const float*)d_in[28], (const float*)d_in[31]};
    const float* gn_b[4] = {(const float*)d_in[23], (const float*)d_in[26],
                            (const float*)d_in[29], (const float*)d_in[32]};
    const float* gn_a[4] = {(const float*)d_in[24], (const float*)d_in[27],
                            (const float*)d_in[30], (const float*)d_in[33]};

    char* ws = (char*)d_ws;
    size_t off = 0;
    auto carve = [&](size_t bytes) -> void* {
        void* p = ws + off;
        off += (bytes + 255) & ~(size_t)255;
        return p;
    };

    _Float16* wt2  = (_Float16*)carve(HH * HH * 2);
    _Float16* wt3  = (_Float16*)carve(HH * HH * 2);
    _Float16* wtg1 = (_Float16*)carve(HH * HH * 2);
    _Float16* wtg2 = (_Float16*)carve(HH * HH * 2);
    _Float16* wtl1 = (_Float16*)carve(HH * HH * 2);
    _Float16* wtl2 = (_Float16*)carve(HH * HH * 2);
    float*    bufA = (float*)carve((size_t)NN * HH * 4);
    float*    bufB = (float*)carve((size_t)NN * HH * 4);
    _Float16* h16a = (_Float16*)carve((size_t)NN * HH * 2);
    _Float16* h16b = (_Float16*)carve((size_t)NN * HH * 2);
    float*    gatebuf = (float*)carve((size_t)NN * 4);
    float*    dinv    = (float*)carve((size_t)NN * 4);
    float*    cnt     = (float*)carve((size_t)BBG * 4);
    float*    meanbuf = (float*)carve((size_t)BBG * HH * 4);
    float*    varbuf  = (float*)carve((size_t)BBG * HH * 4);
    unsigned* m_enc   = (unsigned*)carve((size_t)BBG * 4);
    float*    zbuf    = (float*)carve((size_t)BBG * 4);
    float*    gpool   = (float*)carve((size_t)BBG * HH * 4);
    _Float16* g16a    = (_Float16*)carve((size_t)BBG * HH * 2);
    _Float16* g16b    = (_Float16*)carve((size_t)BBG * HH * 2);

    auto cdiv = [](long a, long b) { return (int)((a + b - 1) / b); };

    // --- weight prep (f16 transposed for WMMA B-fragments) ---
    k_wtrans<<<64, 256, 0, stream>>>(conv2_w, wt2);
    k_wtrans<<<64, 256, 0, stream>>>(conv3_w, wt3);
    k_wtrans<<<64, 256, 0, stream>>>(gate_w1, wtg1);
    k_wtrans<<<64, 256, 0, stream>>>(gate_w2, wtg2);
    k_wtrans<<<64, 256, 0, stream>>>(lin1_w, wtl1);
    k_wtrans<<<64, 256, 0, stream>>>(lin2_w, wtl2);

    // --- per-graph node counts + degree normalization (once) ---
    k_fill<<<cdiv(BBG, 256), 256, 0, stream>>>(cnt, 0.0f, BBG);
    k_count<<<cdiv(NN, 256), 256, 0, stream>>>(batch, cnt, NN);
    k_fill<<<cdiv(NN, 256), 256, 0, stream>>>(dinv, 1.0f, NN);  // self-loop
    k_deg<<<cdiv(EE, 256), 256, 0, stream>>>(dst, dinv, EE);
    k_rsqrt_ip<<<cdiv(NN, 256), 256, 0, stream>>>(dinv, NN);

    auto run_gn = [&](const float* xin, int li, float* outF, _Float16* outH, int C, int relu) {
        int total = NN * C, tb = cdiv(total, 256);
        int bc = BBG * C, bcb = cdiv(bc, 256);
        k_fill<<<bcb, 256, 0, stream>>>(meanbuf, 0.0f, bc);
        k_gn_sum<<<tb, 256, 0, stream>>>(xin, batch, meanbuf, NN, C, relu);
        k_gn_div<<<bcb, 256, 0, stream>>>(meanbuf, cnt, bc, C);
        k_fill<<<bcb, 256, 0, stream>>>(varbuf, 0.0f, bc);
        k_gn_var<<<tb, 256, 0, stream>>>(xin, batch, meanbuf, gn_a[li], varbuf, NN, C, relu);
        k_gn_div<<<bcb, 256, 0, stream>>>(varbuf, cnt, bc, C);
        k_gn_norm<<<tb, 256, 0, stream>>>(xin, batch, meanbuf, varbuf, gn_a[li],
                                          gn_g[li], gn_b[li], outF, outH, NN, C, relu);
    };

    auto run_agg = [&](const float* bias) {  // bufB=xw -> bufA=acc
        k_conv_init<<<cdiv((long)NN * HH, 256), 256, 0, stream>>>(bufB, dinv, bias, bufA, NN);
        k_edge_scatter<<<cdiv((long)EE * 32, 256), 256, 0, stream>>>(bufB, src, dst, dinv, bufA, EE);
    };

    // gn0 on raw x (C=4), fp32 out
    run_gn(x_in, 0, bufA, nullptr, 4, 0);
    // conv1: scalar (K=4), then GCN aggregate
    k_conv1<<<NN, 128, 0, stream>>>(bufA, conv1_w, bufB, NN);
    run_agg(conv1_b);
    // gn1 (relu fused on read) -> f16
    run_gn(bufA, 1, nullptr, h16a, HH, 1);
    // conv2: WMMA GEMM -> xw, then aggregate
    k_gemm_wmma<<<cdiv(NN, 128), 256, 0, stream>>>(h16a, wt2, nullptr, bufB, nullptr, NN, 0);
    run_agg(conv2_b);
    // gn2 -> f16
    run_gn(bufA, 2, nullptr, h16a, HH, 1);
    // conv3
    k_gemm_wmma<<<cdiv(NN, 128), 256, 0, stream>>>(h16a, wt3, nullptr, bufB, nullptr, NN, 0);
    run_agg(conv3_b);
    // gn3 -> f32 (for pooling) + f16 (for gate MLP)
    run_gn(bufA, 3, bufB, h16a, HH, 1);

    // gate MLP (WMMA, fused bias+relu, f16 chaining)
    k_gemm_wmma<<<cdiv(NN, 128), 256, 0, stream>>>(h16a, wtg1, gate_b1, nullptr, h16b, NN, 1);
    k_gemm_wmma<<<cdiv(NN, 128), 256, 0, stream>>>(h16b, wtg2, gate_b2, nullptr, h16a, NN, 1);
    k_dot_h<<<cdiv(NN, 256), 256, 0, stream>>>(h16a, gate_w3, gate_b3, gatebuf, NN);

    // softmax-per-graph + weighted pooling
    k_fill<<<cdiv(BBG, 256), 256, 0, stream>>>((float*)m_enc, 0.0f, BBG);
    k_fill<<<cdiv(BBG, 256), 256, 0, stream>>>(zbuf, 0.0f, BBG);
    k_fill<<<cdiv(BBG * HH, 256), 256, 0, stream>>>(gpool, 0.0f, BBG * HH);
    k_segmax<<<cdiv(NN, 256), 256, 0, stream>>>(gatebuf, batch, m_enc, NN);
    k_expsum<<<cdiv(NN, 256), 256, 0, stream>>>(gatebuf, batch, m_enc, zbuf, NN);
    k_pool<<<cdiv((long)NN * 32, 256), 256, 0, stream>>>(gatebuf, zbuf, batch, bufB, gpool, NN);

    // head MLP on [512,128] (WMMA)
    k_f32h<<<cdiv(BBG * HH, 256), 256, 0, stream>>>(gpool, g16a, BBG * HH);
    k_gemm_wmma<<<cdiv(BBG, 128), 256, 0, stream>>>(g16a, wtl1, lin1_b, nullptr, g16b, BBG, 1);
    k_gemm_wmma<<<cdiv(BBG, 128), 256, 0, stream>>>(g16b, wtl2, lin2_b, nullptr, g16a, BBG, 1);
    k_dot_h<<<cdiv(BBG, 256), 256, 0, stream>>>(g16a, last_w, last_b, (float*)d_out, BBG);
}
